// GAT_75969381531755
// MI455X (gfx1250) — compile-verified, matrix operand order checked by reading, not run on hardware
//
#include <hip/hip_runtime.h>
#include <hip/hip_bf16.h>

// -------------------------------------------------------------------------
// 2-layer single-head GATConv (PyG semantics, add_self_loops=False) for
// MI455X / gfx1250.
//   * Dense projections: v_wmma_f32_16x16x32_f16 (f16 in, f32 accumulate).
//   * Weight operand staged into LDS via the Tensor Data Mover
//     (tensor_load_to_lds + s_wait_tensorcnt), descriptor built per the
//     CDNA5 ISA D# layout (group0/group1 bitfields).
//   * Edge softmax/scatter phase is L2-resident and atomics-based.
// -------------------------------------------------------------------------

typedef __attribute__((ext_vector_type(16))) _Float16 v16h;
typedef __attribute__((ext_vector_type(8)))  float    v8f;

#if defined(__gfx1250__) && __has_builtin(__builtin_amdgcn_tensor_load_to_lds)
#define GAT_HAS_TDM 1
typedef unsigned int gat_v4u __attribute__((ext_vector_type(4)));
typedef int          gat_v8i __attribute__((ext_vector_type(8)));
typedef int          gat_v4i __attribute__((ext_vector_type(4)));

// Issue one TDM load of a [rows x 64] f32 row-major tile (global -> LDS)
// and wait for completion.  Descriptor per CDNA5 ISA §8 (Tensor DMA D#):
//   group0: count=1 | lds_addr | global_addr[56:0] | type=2
//   group1: data_size=4B, tensor_dim0=64, tensor_dim1=rows,
//           tile_dim0=64, tile_dim1=rows, tensor_dim0_stride=64
__device__ __forceinline__ void gat_tdm_load_w(const float* gsrc, float* lds_dst,
                                               unsigned rows) {
    typedef __attribute__((address_space(3))) float lds_float;
    lds_float* l3 = (lds_float*)lds_dst;                 // generic -> LDS offset
    const unsigned lds_off = (unsigned)(unsigned long long)l3;
    const unsigned long long ga = (unsigned long long)gsrc;

    gat_v4u g0;
    g0[0] = 1u;                                          // count=1, user D#
    g0[1] = lds_off;                                     // lds_addr
    g0[2] = (unsigned)(ga & 0xFFFFFFFFull);              // global_addr[31:0]
    g0[3] = (unsigned)((ga >> 32) & 0x01FFFFFFull)       // global_addr[56:32]
            | 0x80000000u;                               // type=2 ("image")

    gat_v8i g1;
    g1[0] = (int)(2u << 16);                             // data_size=2 -> 4 bytes
    g1[1] = (int)(64u << 16);                            // tensor_dim0[15:0]=64
    g1[2] = (int)(rows << 16);                           // dim0[31:16]=0 | tensor_dim1[15:0]=rows
    g1[3] = (int)(64u << 16);                            // tensor_dim1[31:16]=0 | tile_dim0=64
    g1[4] = (int)rows;                                   // tile_dim1=rows | tile_dim2=0
    g1[5] = 64;                                          // tensor_dim0_stride[31:0]=64
    g1[6] = 0;                                           // stride0 hi | stride1 lo
    g1[7] = 0;                                           // stride1 hi

    gat_v4i z4 = {0, 0, 0, 0};
#if __clang_major__ >= 23
    gat_v8i z8 = {0, 0, 0, 0, 0, 0, 0, 0};
    __builtin_amdgcn_tensor_load_to_lds(g0, g1, z4, z4, z8, 0);
#else
    __builtin_amdgcn_tensor_load_to_lds(g0, g1, z4, z4, 0);
#endif
    __builtin_amdgcn_s_wait_tensorcnt(0);
}
#endif

// Monotonic float <-> uint mapping so segment-max can use integer atomicMax
// (order-independent => deterministic).
__device__ __forceinline__ unsigned gat_ford(float f) {
    unsigned u = __float_as_uint(f);
    return (u & 0x80000000u) ? ~u : (u | 0x80000000u);
}
__device__ __forceinline__ float gat_funord(unsigned u) {
    unsigned b = (u & 0x80000000u) ? (u & 0x7FFFFFFFu) : ~u;
    return __uint_as_float(b);
}

// ---------------- WMMA GEMM:  C[N,64] = A[N,K] @ B[K,64]  -----------------
// Block = 256 threads = 8 waves. Wave w computes tile
//   rows [blockIdx.x*32 + (w>>2)*16, +16), cols [(w&3)*16, +16).
// A-fragment layout (16-bit A 16x32, ISA 7.12.2): lane holds row M=lane&15,
// K-halves selected by lane[4]; B laid out symmetrically (lane = col N&15).
// B ([K,64] f32) is staged once per block into LDS via the TDM.
template <int K>
__global__ void gat_gemm(const float* __restrict__ A,
                         const float* __restrict__ B,   // [K,64] row-major
                         float* __restrict__ C,         // [N,64]
                         int nrows) {
    __shared__ float Bs[K * 64];

    const int lane = threadIdx.x & 31;
    const int wave = threadIdx.x >> 5;

#if GAT_HAS_TDM
    if (wave == 0) {
        gat_tdm_load_w(B, Bs, (unsigned)K);
    }
#else
    for (int i = threadIdx.x; i < K * 64; i += 256) Bs[i] = B[i];
#endif
    __syncthreads();

    const int row0 = blockIdx.x * 32 + (wave >> 2) * 16;
    const int col0 = (wave & 3) * 16;
    const int m    = lane & 15;
    const int kh   = (lane >> 4) * 8;       // 0 or 8: K-half select

    int arow = row0 + m;
    if (arow >= nrows) arow = nrows - 1;    // clamp loads; stores are guarded
    const float2* __restrict__ A2 = (const float2*)(A + (size_t)arow * K);

    v8f acc = {};
#pragma unroll
    for (int kb = 0; kb < K; kb += 32) {
        v16h af, bf;
#pragma unroll
        for (int p = 0; p < 8; ++p) {
            const int k = kb + ((p & 4) << 2) + kh + (p & 3) * 2;  // (p<4?0:16)+kh+(p&3)*2
            const float2 av = A2[k >> 1];                          // global_load_b64
            af[2 * p]     = (_Float16)av.x;
            af[2 * p + 1] = (_Float16)av.y;
            bf[2 * p]     = (_Float16)Bs[k * 64 + col0 + m];       // ds_load
            bf[2 * p + 1] = (_Float16)Bs[(k + 1) * 64 + col0 + m];
        }
        acc = __builtin_amdgcn_wmma_f32_16x16x32_f16(
            /*neg_a=*/false, af, /*neg_b=*/false, bf,
            /*c_mod=*/(short)0, acc, /*reuse_a=*/false, /*reuse_b=*/false);
    }

    // C/D layout: VGPR r -> (M=r, N=lane) for lanes 0-15, (M=r+8) for 16-31.
    const int rbase = row0 + (lane >> 4) * 8;
    float* cp = C + (size_t)rbase * 64 + (col0 + m);
    if (rbase + 7 < nrows) {
#pragma unroll
        for (int r = 0; r < 8; ++r) cp[(size_t)r * 64] = acc[r];   // immediate offsets
    } else {
#pragma unroll
        for (int r = 0; r < 8; ++r)
            if (rbase + r < nrows) cp[(size_t)r * 64] = acc[r];
    }
}

// -------- per-node attention scores: alpha_s = h.a_src, alpha_d = h.a_dst --
__global__ void gat_attn(const float* __restrict__ h,
                         const float* __restrict__ asrc,
                         const float* __restrict__ adst,
                         float* __restrict__ out_s,
                         float* __restrict__ out_d, int n) {
    const int node = (int)(((size_t)blockIdx.x * blockDim.x + threadIdx.x) >> 5);
    const int lane = threadIdx.x & 31;
    if (node >= n) return;
    const float* hp = h + (size_t)node * 64;
    float h0 = hp[lane], h1 = hp[lane + 32];
    float s = h0 * asrc[lane] + h1 * asrc[lane + 32];
    float d = h0 * adst[lane] + h1 * adst[lane + 32];
#pragma unroll
    for (int off = 16; off; off >>= 1) {
        s += __shfl_xor(s, off);
        d += __shfl_xor(d, off);
    }
    if (lane == 0) { out_s[node] = s; out_d[node] = d; }
}

// -------- edge pass A: e = leakyReLU(as[src]+ad[dst]); segment max ---------
__global__ void gat_edge_scores(const int* __restrict__ src,
                                const int* __restrict__ dst,
                                const float* __restrict__ as,
                                const float* __restrict__ ad,
                                float* __restrict__ ebuf,
                                unsigned* __restrict__ mu, int ne) {
    const int i = blockIdx.x * blockDim.x + threadIdx.x;
    if (i >= ne) return;
    float e = as[src[i]] + ad[dst[i]];
    e = (e > 0.f) ? e : 0.2f * e;          // negative_slope = 0.2
    ebuf[i] = e;
    atomicMax(&mu[dst[i]], gat_ford(e));
}

// -------- edge pass B: w = exp(e - m[dst]); segment sum --------------------
__global__ void gat_edge_exp(const int* __restrict__ dst,
                             const float* __restrict__ ebuf,
                             const unsigned* __restrict__ mu,
                             float* __restrict__ wbuf,
                             float* __restrict__ z, int ne) {
    const int i = blockIdx.x * blockDim.x + threadIdx.x;
    if (i >= ne) return;
    const int d = dst[i];
    const float w = expf(ebuf[i] - gat_funord(mu[d]));
    wbuf[i] = w;
    atomicAdd(&z[d], w);
}

// -------- edge pass C (wave per edge): agg[dst] += h[src] * (w/z[dst]) -----
__global__ void gat_edge_agg(const int* __restrict__ src,
                             const int* __restrict__ dst,
                             const float* __restrict__ wbuf,
                             const float* __restrict__ z,
                             const float* __restrict__ h,
                             float* __restrict__ agg, int ne) {
    const int i = (int)(((size_t)blockIdx.x * blockDim.x + threadIdx.x) >> 5);
    const int lane = threadIdx.x & 31;
    if (i >= ne) return;
    const int d = dst[i], s = src[i];
    const float alpha = wbuf[i] / z[d];
    const float* hp = h + (size_t)s * 64;
    float* op = agg + (size_t)d * 64;
    atomicAdd(&op[lane],      hp[lane]      * alpha);
    atomicAdd(&op[lane + 32], hp[lane + 32] * alpha);
}

// -------- layer-1 epilogue: out = relu(agg + b) (elementwise) --------------
__global__ void gat_bias_relu(const float* __restrict__ agg,
                              const float* __restrict__ b,
                              float* __restrict__ out, int n64) {
    const int i = blockIdx.x * blockDim.x + threadIdx.x;
    if (i >= n64) return;
    const float v = agg[i] + b[i & 63];
    out[i] = (v > 0.f) ? v : 0.f;
}

// -------- layer-2 epilogue: out = (agg + b) / max(||agg+b||, 1e-12) --------
__global__ void gat_bias_l2norm(const float* __restrict__ agg,
                                const float* __restrict__ b,
                                float* __restrict__ out, int n) {
    const int node = (int)(((size_t)blockIdx.x * blockDim.x + threadIdx.x) >> 5);
    const int lane = threadIdx.x & 31;
    if (node >= n) return;
    const size_t base = (size_t)node * 64;
    float v0 = agg[base + lane]      + b[lane];
    float v1 = agg[base + lane + 32] + b[lane + 32];
    float ss = v0 * v0 + v1 * v1;
#pragma unroll
    for (int off = 16; off; off >>= 1) ss += __shfl_xor(ss, off);
    const float inv = 1.f / fmaxf(sqrtf(ss), 1e-12f);
    out[base + lane]      = v0 * inv;
    out[base + lane + 32] = v1 * inv;
}

// -------------------------------------------------------------------------
extern "C" void kernel_launch(void* const* d_in, const int* in_sizes, int n_in,
                              void* d_out, int out_size, void* d_ws, size_t ws_size,
                              hipStream_t stream) {
    const float* x     = (const float*)d_in[0];
    const int*   ei    = (const int*)d_in[1];      // [2, E]
    const float* W1    = (const float*)d_in[2];    // [128,64]
    const float* asrc1 = (const float*)d_in[3];
    const float* adst1 = (const float*)d_in[4];
    const float* b1    = (const float*)d_in[5];
    const float* W2    = (const float*)d_in[6];    // [64,64]
    const float* asrc2 = (const float*)d_in[7];
    const float* adst2 = (const float*)d_in[8];
    const float* b2    = (const float*)d_in[9];

    const int N = in_sizes[0] / 128;
    const int E = in_sizes[1] / 2;
    const int* src = ei;
    const int* dst = ei + E;

    // --- carve workspace ---
    char* p = (char*)d_ws;
    auto carve = [&](size_t bytes) -> void* {
        void* r = (void*)p;
        p += (bytes + 255) & ~(size_t)255;
        return r;
    };
    float*    hbuf   = (float*)carve((size_t)N * 64 * sizeof(float));
    float*    aggbuf = (float*)carve((size_t)N * 64 * sizeof(float));
    float*    ebuf   = (float*)carve((size_t)E * sizeof(float));
    float*    wbuf   = (float*)carve((size_t)E * sizeof(float));
    float*    as     = (float*)carve((size_t)N * sizeof(float));
    float*    ad     = (float*)carve((size_t)N * sizeof(float));
    float*    z      = (float*)carve((size_t)N * sizeof(float));
    unsigned* mu     = (unsigned*)carve((size_t)N * sizeof(unsigned));

    const dim3 blk(256);
    const int gemm_blocks     = (N + 31) / 32;
    const int nodewave_blocks = (int)(((size_t)N * 32 + 255) / 256);
    const int edge_blocks     = (E + 255) / 256;
    const int edgewave_blocks = (int)(((size_t)E * 32 + 255) / 256);
    const int elem_blocks     = (int)(((size_t)N * 64 + 255) / 256);

    // ================= layer 1 =================
    gat_gemm<128><<<gemm_blocks, blk, 0, stream>>>(x, W1, hbuf, N);
    gat_attn<<<nodewave_blocks, blk, 0, stream>>>(hbuf, asrc1, adst1, as, ad, N);
    hipMemsetAsync(mu, 0, (size_t)N * sizeof(unsigned), stream);
    hipMemsetAsync(z, 0, (size_t)N * sizeof(float), stream);
    hipMemsetAsync(aggbuf, 0, (size_t)N * 64 * sizeof(float), stream);
    gat_edge_scores<<<edge_blocks, blk, 0, stream>>>(src, dst, as, ad, ebuf, mu, E);
    gat_edge_exp<<<edge_blocks, blk, 0, stream>>>(dst, ebuf, mu, wbuf, z, E);
    gat_edge_agg<<<edgewave_blocks, blk, 0, stream>>>(src, dst, wbuf, z, hbuf, aggbuf, E);
    gat_bias_relu<<<elem_blocks, blk, 0, stream>>>(aggbuf, b1, aggbuf, N * 64);

    // ================= layer 2 =================
    float* out = (float*)d_out;
    gat_gemm<64><<<gemm_blocks, blk, 0, stream>>>(aggbuf, W2, hbuf, N);
    gat_attn<<<nodewave_blocks, blk, 0, stream>>>(hbuf, asrc2, adst2, as, ad, N);
    hipMemsetAsync(mu, 0, (size_t)N * sizeof(unsigned), stream);
    hipMemsetAsync(z, 0, (size_t)N * sizeof(float), stream);
    hipMemsetAsync(out, 0, (size_t)N * 64 * sizeof(float), stream);
    gat_edge_scores<<<edge_blocks, blk, 0, stream>>>(src, dst, as, ad, ebuf, mu, E);
    gat_edge_exp<<<edge_blocks, blk, 0, stream>>>(dst, ebuf, mu, wbuf, z, E);
    gat_edge_agg<<<edgewave_blocks, blk, 0, stream>>>(src, dst, wbuf, z, hbuf, out, E);
    gat_bias_l2norm<<<nodewave_blocks, blk, 0, stream>>>(out, b2, out, N);
}
